// Block_67173288509603
// MI455X (gfx1250) — matrix operand order repaired and checked by reading, not compiled, not run on hardware
//
#include <hip/hip_runtime.h>
#include <hip/hip_bf16.h>
#include <math.h>

// ---------------- problem constants ----------------
#define B_    2
#define T_    2048
#define C_    1024
#define H_    16
#define KVH_  4
#define HD_   64
#define GROUPS_ 4
#define BLK_  128
#define E_    8
#define SH_H_ 2048
#define RE_H_ 1024
#define NTOK  (B_ * T_)          // 4096

// ---------------- WMMA types (probe-verified for gfx1250) ----------------
typedef __attribute__((ext_vector_type(16))) __bf16 v16bf;
typedef __attribute__((ext_vector_type(8)))  float  v8f;

union FragBF {
    v16bf v;
    unsigned short h[16];
    uint4 q[2];
};

__device__ __forceinline__ unsigned short f2bf(float f) {
    union { float f; unsigned u; } c; c.f = f;
    unsigned u = c.u;
    u += 0x7FFFu + ((u >> 16) & 1u);   // RNE
    return (unsigned short)(u >> 16);
}

__device__ __forceinline__ v8f wmma_bf16(const FragBF& a, const FragBF& b, v8f c) {
    return __builtin_amdgcn_wmma_f32_16x16x32_bf16(false, a.v, false, b.v,
                                                   (short)0, c, false, false);
}

__device__ __forceinline__ FragBF load_bfrag(const unsigned short* p) {
    FragBF f;
    f.q[0] = ((const uint4*)p)[0];
    f.q[1] = ((const uint4*)p)[1];
    return f;
}

// A fragment from a bf16 row-major [M,K] matrix: two 16B loads, no VALU
__device__ __forceinline__ FragBF load_afrag_bf16(const unsigned short* Arow,
                                                  int k0, int hi) {
    FragBF a;
    a.q[0] = *(const uint4*)(Arow + k0 + hi * 16);
    a.q[1] = *(const uint4*)(Arow + k0 + hi * 16 + 8);
    return a;
}

// =====================================================================
// Weight convert: f32 [K,N] -> bf16 [N,K] (transposed), batched over z
// =====================================================================
__global__ __launch_bounds__(256)
void w_to_bf16t(const float* __restrict__ in, unsigned short* __restrict__ out,
                int K, int N) {
    int z = blockIdx.z;
    in  += (size_t)z * K * N;
    out += (size_t)z * K * N;
    __shared__ float tile[32][33];
    int x  = blockIdx.x * 32 + threadIdx.x;   // N index
    int y0 = blockIdx.y * 32;                 // K base
#pragma unroll
    for (int i = threadIdx.y; i < 32; i += 8)
        tile[i][threadIdx.x] = in[(size_t)(y0 + i) * N + x];
    __syncthreads();
    int xo = y0 + threadIdx.x;                // K index in output row
#pragma unroll
    for (int i = threadIdx.y; i < 32; i += 8)
        out[(size_t)(blockIdx.x * 32 + i) * K + xo] = f2bf(tile[threadIdx.x][i]);
}

// =====================================================================
// AdaLN modulation: out[B,2C] = t_emb[B,C] @ w[C,2C] + b
// =====================================================================
__global__ __launch_bounds__(256)
void timemod_kernel(const float* __restrict__ t_emb, const float* __restrict__ w,
                    const float* __restrict__ bias, float* __restrict__ out) {
    int i = blockIdx.x * 256 + threadIdx.x;   // B*2C = 4096
    int b = i >> 11;
    int j = i & 2047;
    float acc = bias[j];
    const float* te = t_emb + (size_t)b * C_;
    for (int k = 0; k < C_; k++) acc += te[k] * w[(size_t)k * 2048 + j];
    out[i] = acc;
}

// =====================================================================
// AdaLN: LN(x)*(1+scale)+shift; writes f32 (router) AND bf16 (GEMM A)
// =====================================================================
__global__ __launch_bounds__(256)
void adaln_kernel(const float* __restrict__ x, const float* __restrict__ ss,
                  float* __restrict__ outf, unsigned short* __restrict__ outb) {
    __shared__ float red[256];
    int n = blockIdx.x;
    int tid = threadIdx.x;
    const float* row = x + (size_t)n * C_;
    float v0 = row[tid], v1 = row[tid + 256], v2 = row[tid + 512], v3 = row[tid + 768];
    red[tid] = v0 + v1 + v2 + v3;
    __syncthreads();
    for (int o = 128; o > 0; o >>= 1) { if (tid < o) red[tid] += red[tid + o]; __syncthreads(); }
    float mean = red[0] * (1.0f / 1024.0f);
    __syncthreads();
    float d0 = v0 - mean, d1 = v1 - mean, d2 = v2 - mean, d3 = v3 - mean;
    red[tid] = d0 * d0 + d1 * d1 + d2 * d2 + d3 * d3;
    __syncthreads();
    for (int o = 128; o > 0; o >>= 1) { if (tid < o) red[tid] += red[tid + o]; __syncthreads(); }
    float rs = rsqrtf(red[0] * (1.0f / 1024.0f) + 1e-5f);
    int b = n >> 11;  // n / T_
    const float* sc = ss + (size_t)b * 2048;
    const float* sh = sc + 1024;
    float* orow = outf + (size_t)n * C_;
    unsigned short* brow = outb + (size_t)n * C_;
    float r0 = d0 * rs * (1.0f + sc[tid])       + sh[tid];
    float r1 = d1 * rs * (1.0f + sc[tid + 256]) + sh[tid + 256];
    float r2 = d2 * rs * (1.0f + sc[tid + 512]) + sh[tid + 512];
    float r3 = d3 * rs * (1.0f + sc[tid + 768]) + sh[tid + 768];
    orow[tid] = r0;  orow[tid + 256] = r1;  orow[tid + 512] = r2;  orow[tid + 768] = r3;
    brow[tid] = f2bf(r0);           brow[tid + 256] = f2bf(r1);
    brow[tid + 512] = f2bf(r2);     brow[tid + 768] = f2bf(r3);
}

// =====================================================================
// Generic WMMA GEMM: Out[M,N] = A[M,K](bf16) * W[N,K](bf16, transposed)
//   - 8 waves/block; wave = 32x64 tile (2 A frags share each B frag,
//     8 WMMA per k-step; hot loop = 12 b128 loads : 8 wmma, no VALU)
//   - compile-time: ACT (0 none / 1 exact GELU), RESID (f32 residual),
//     GUARD (MoE dynamic-M via counts/offsets), GATHER (rowmap), OBF
//     (bf16 output instead of f32)
// =====================================================================
template<int ACT, bool RESID, bool GUARD, bool GATHER, bool OBF>
__global__ __launch_bounds__(256)
void wmma_gemm_t(const unsigned short* __restrict__ A,
                 const unsigned short* __restrict__ W,
                 void* __restrict__ OutP, const float* __restrict__ resid,
                 const int* __restrict__ rowmap, const int* __restrict__ counts,
                 const int* __restrict__ offsets,
                 int M, int N, int K, long long wstride) {
    int z = blockIdx.z;
    int cnt = GUARD ? counts[z] : M;
    int off = GUARD ? offsets[z] : 0;
    if (GUARD && (int)(blockIdx.y * 256) >= cnt) return;
    const unsigned short* Wz = W + (size_t)z * (size_t)wstride;
    const int* rmap = GATHER ? (rowmap + (size_t)z * NTOK) : nullptr;

    int lane = threadIdx.x & 31;
    int wave = threadIdx.x >> 5;
    int hi = lane >> 4;
    int lo = lane & 15;
    int m0 = blockIdx.y * 256 + wave * 32;
    int n0 = blockIdx.x * 64;

    // A rows for the two 16-row fragments
    int rl0 = m0 + lo;
    int rl1 = m0 + 16 + lo;
    int arow0, arow1;
    if (GUARD) {
        if (GATHER) {
            arow0 = (rl0 < cnt) ? rmap[rl0] : 0;
            arow1 = (rl1 < cnt) ? rmap[rl1] : 0;
        } else {
            arow0 = (rl0 < cnt) ? (off + rl0) : off;
            arow1 = (rl1 < cnt) ? (off + rl1) : off;
        }
    } else {
        arow0 = rl0;
        arow1 = rl1;
    }
    const unsigned short* Arow0 = A + (size_t)arow0 * K;
    const unsigned short* Arow1 = A + (size_t)arow1 * K;

    v8f acc0[4], acc1[4];
#pragma unroll
    for (int nt = 0; nt < 4; nt++) { acc0[nt] = (v8f){}; acc1[nt] = (v8f){}; }

    for (int k0 = 0; k0 < K; k0 += 32) {
        FragBF a0 = load_afrag_bf16(Arow0, k0, hi);
        FragBF a1 = load_afrag_bf16(Arow1, k0, hi);
#pragma unroll
        for (int nt = 0; nt < 4; nt++) {
            const unsigned short* wp =
                Wz + (size_t)(n0 + nt * 16 + lo) * K + k0 + hi * 16;
            FragBF bf = load_bfrag(wp);
            acc0[nt] = wmma_bf16(a0, bf, acc0[nt]);
            acc1[nt] = wmma_bf16(a1, bf, acc1[nt]);
        }
    }

    float* Outf = (float*)OutP;
    unsigned short* Outb = (unsigned short*)OutP;
#pragma unroll
    for (int half = 0; half < 2; half++) {
        v8f* acc = half ? acc1 : acc0;
        int mbase = m0 + half * 16;
#pragma unroll
        for (int nt = 0; nt < 4; nt++) {
            int col = n0 + nt * 16 + lo;
#pragma unroll
            for (int j = 0; j < 8; j++) {
                int ml = mbase + j + hi * 8;
                if (!GUARD || ml < cnt) {
                    size_t orow = (size_t)(off + ml);
                    float v = acc[nt][j];
                    if (RESID) v += resid[orow * N + col];
                    if (ACT == 1) v = 0.5f * v * (1.0f + erff(v * 0.70710678118f));
                    if (OBF) Outb[orow * N + col] = f2bf(v);
                    else     Outf[orow * N + col] = v;
                }
            }
        }
    }
}

// =====================================================================
// RoPE + bf16 pack
// =====================================================================
__global__ __launch_bounds__(256)
void rope_pack_q(const float* __restrict__ xq, unsigned short* __restrict__ qbf) {
    int i = blockIdx.x * 256 + threadIdx.x;    // B*T*H*HD = 4M
    int d = i & 63;
    int h = (i >> 6) & 15;
    int t = (i >> 10) & 2047;
    int b = i >> 21;
    const float* row = xq + ((size_t)(b * T_ + t)) * (H_ * HD_) + h * HD_;
    float x = row[d];
    int fi = d & 31;
    float ang = (float)t * __expf(-0.28782313662f * (float)fi);  // ln(1e4)*2/64
    float cv = cosf(ang), sv = sinf(ang);
    float rot = (d < 32) ? -row[d + 32] : row[d - 32];
    qbf[((size_t)((b * H_ + h) * T_) + t) * HD_ + d] = f2bf(x * cv + rot * sv);
}

__global__ __launch_bounds__(256)
void rope_pack_k(const float* __restrict__ xk, unsigned short* __restrict__ kbf) {
    int i = blockIdx.x * 256 + threadIdx.x;    // B*T*KVH*HD = 1M
    int d = i & 63;
    int kv = (i >> 6) & 3;
    int t = (i >> 8) & 2047;
    int b = i >> 19;
    const float* row = xk + ((size_t)(b * T_ + t)) * (KVH_ * HD_) + kv * HD_;
    float x = row[d];
    int fi = d & 31;
    float ang = (float)t * __expf(-0.28782313662f * (float)fi);
    float cv = cosf(ang), sv = sinf(ang);
    float rot = (d < 32) ? -row[d + 32] : row[d - 32];
    kbf[((size_t)((b * KVH_ + kv) * T_) + t) * HD_ + d] = f2bf(x * cv + rot * sv);
}

__global__ __launch_bounds__(256)
void v_pack_t(const float* __restrict__ xv, unsigned short* __restrict__ vt) {
    int i = blockIdx.x * 256 + threadIdx.x;    // B*T*KVH*HD = 1M
    int d = i & 63;
    int kv = (i >> 6) & 3;
    int t = (i >> 8) & 2047;
    int b = i >> 19;
    float x = xv[((size_t)(b * T_ + t)) * (KVH_ * HD_) + kv * HD_ + d];
    vt[((size_t)((b * KVH_ + kv) * HD_) + d) * T_ + t] = f2bf(x);
}

// =====================================================================
// Flash attention (block-causal @128, GQA): 1 wave per 16-query tile
//   S = Q*K^T (2 WMMA chained over d), online softmax, O += P*V (WMMA)
//   Output written as bf16 (feeds the wo GEMM directly)
// =====================================================================
__global__ __launch_bounds__(32)
void flash_attn(const unsigned short* __restrict__ qbf,   // [B,H,T,HD]
                const unsigned short* __restrict__ kbf,   // [B,KVH,T,HD]
                const unsigned short* __restrict__ vtb,   // [B,KVH,HD,T]
                unsigned short* __restrict__ y) {         // [B*T, H*HD] bf16
    __shared__ unsigned short pst[16 * 40];               // P restage, padded
    int lane = threadIdx.x;
    int hi = lane >> 4, lo = lane & 15;
    int qt = blockIdx.x, h = blockIdx.y, b = blockIdx.z;
    int kv = h / GROUPS_;
    int q0 = qt * 16;
    int kmax = ((q0 / BLK_) + 1) * BLK_;                  // block-causal limit

    const unsigned short* qbase = qbf + ((size_t)(b * H_ + h)) * T_ * HD_;
    const unsigned short* kbase = kbf + ((size_t)(b * KVH_ + kv)) * T_ * HD_;
    const unsigned short* vbase = vtb + ((size_t)(b * KVH_ + kv)) * HD_ * T_;

    FragBF qa0, qa1;                                      // Q 16x64 as two A frags
    {
        const unsigned short* qr = qbase + (size_t)(q0 + lo) * HD_;
        qa0.q[0] = *(const uint4*)(qr + hi * 16);
        qa0.q[1] = *(const uint4*)(qr + hi * 16 + 8);
        qa1.q[0] = *(const uint4*)(qr + 32 + hi * 16);
        qa1.q[1] = *(const uint4*)(qr + 40 + hi * 16);
    }

    v8f o0 = {}, o1 = {}, o2 = {}, o3 = {};
    float mrow[8], lrow[8];
#pragma unroll
    for (int j = 0; j < 8; j++) { mrow[j] = -3.0e38f; lrow[j] = 0.0f; }

    for (int kb = 0; kb < kmax; kb += 32) {
        v8f s0 = {}, s1 = {};
        {   // key tile 0 (keys kb..kb+15), B frags over d
            const unsigned short* kr = kbase + (size_t)(kb + lo) * HD_ + hi * 16;
            FragBF f0 = load_bfrag(kr);
            FragBF f1 = load_bfrag(kr + 32);
            s0 = wmma_bf16(qa0, f0, s0);
            s0 = wmma_bf16(qa1, f1, s0);
        }
        {   // key tile 1 (keys kb+16..kb+31)
            const unsigned short* kr = kbase + (size_t)(kb + 16 + lo) * HD_ + hi * 16;
            FragBF f0 = load_bfrag(kr);
            FragBF f1 = load_bfrag(kr + 32);
            s1 = wmma_bf16(qa0, f0, s1);
            s1 = wmma_bf16(qa1, f1, s1);
        }
        // online softmax per row; rows live in 16-lane groups (xor<=8 stays inside)
#pragma unroll
        for (int j = 0; j < 8; j++) {
            float a0 = s0[j] * 0.125f, a1 = s1[j] * 0.125f;
            float mx = fmaxf(a0, a1);
#pragma unroll
            for (int dd = 1; dd < 16; dd <<= 1) mx = fmaxf(mx, __shfl_xor(mx, dd, 32));
            float mnew = fmaxf(mrow[j], mx);
            float alpha = __expf(mrow[j] - mnew);
            float p0 = __expf(a0 - mnew), p1 = __expf(a1 - mnew);
            float rs = p0 + p1;
#pragma unroll
            for (int dd = 1; dd < 16; dd <<= 1) rs += __shfl_xor(rs, dd, 32);
            lrow[j] = lrow[j] * alpha + rs;
            mrow[j] = mnew;
            o0[j] *= alpha; o1[j] *= alpha; o2[j] *= alpha; o3[j] *= alpha;
            int m = j + hi * 8;
            pst[m * 40 + lo]      = f2bf(p0);
            pst[m * 40 + 16 + lo] = f2bf(p1);
        }
        __syncthreads();
        FragBF pa;    // P 16x32 as A fragment from LDS
        {
            const unsigned short* pr = pst + lo * 40;
            pa.q[0] = *(const uint4*)(pr + hi * 16);
            pa.q[1] = *(const uint4*)(pr + hi * 16 + 8);
        }
        __syncthreads();
        {   // O += P * V ; V^T layout -> contiguous B frags over t
            const unsigned short* v0p = vbase + (size_t)(0 * 16 + lo) * T_ + kb + hi * 16;
            const unsigned short* v1p = vbase + (size_t)(1 * 16 + lo) * T_ + kb + hi * 16;
            const unsigned short* v2p = vbase + (size_t)(2 * 16 + lo) * T_ + kb + hi * 16;
            const unsigned short* v3p = vbase + (size_t)(3 * 16 + lo) * T_ + kb + hi * 16;
            FragBF f;
            f = load_bfrag(v0p); o0 = wmma_bf16(pa, f, o0);
            f = load_bfrag(v1p); o1 = wmma_bf16(pa, f, o1);
            f = load_bfrag(v2p); o2 = wmma_bf16(pa, f, o2);
            f = load_bfrag(v3p); o3 = wmma_bf16(pa, f, o3);
        }
    }
#pragma unroll
    for (int j = 0; j < 8; j++) {
        float inv = 1.0f / lrow[j];
        int q = q0 + j + hi * 8;
        unsigned short* yr = y + ((size_t)(b * T_ + q)) * (H_ * HD_) + h * HD_;
        yr[lo]      = f2bf(o0[j] * inv);
        yr[16 + lo] = f2bf(o1[j] * inv);
        yr[32 + lo] = f2bf(o2[j] * inv);
        yr[48 + lo] = f2bf(o3[j] * inv);
    }
}

// =====================================================================
// SwiGLU combine: mid = silu(a) * b, written as bf16 (feeds sw2 GEMM)
// =====================================================================
__global__ __launch_bounds__(256)
void silu_mul(const float* __restrict__ a, const float* __restrict__ b,
              unsigned short* __restrict__ mid) {
    int i = blockIdx.x * 256 + threadIdx.x;
    float x = a[i];
    mid[i] = f2bf((x / (1.0f + __expf(-x))) * b[i]);
}

// =====================================================================
// Router: s[n,e] = sigmoid(h2[n] . router_w[:,e])
// =====================================================================
__global__ __launch_bounds__(256)
void router_scores(const float* __restrict__ h2, const float* __restrict__ rw,
                   float* __restrict__ s) {
    int i = blockIdx.x * 256 + threadIdx.x;   // NTOK*E
    int n = i >> 3;
    int e = i & 7;
    const float* hr = h2 + (size_t)n * C_;
    float acc = 0.0f;
    for (int k = 0; k < C_; k++) acc += hr[k] * rw[(size_t)k * E_ + e];
    s[i] = 1.0f / (1.0f + __expf(-acc));
}

// Top-2 (ties -> lowest index, matching lax.top_k) + gate normalization
__global__ __launch_bounds__(256)
void topk_kernel(const float* __restrict__ s, const float* __restrict__ bias,
                 int* __restrict__ idx, float* __restrict__ gates) {
    int n = blockIdx.x * 256 + threadIdx.x;
    if (n >= NTOK) return;
    float sv[E_], sel[E_];
#pragma unroll
    for (int e = 0; e < E_; e++) { sv[e] = s[n * E_ + e]; sel[e] = sv[e] + bias[e]; }
    int i1 = 0; float b1 = sel[0];
#pragma unroll
    for (int e = 1; e < E_; e++) if (sel[e] > b1) { b1 = sel[e]; i1 = e; }
    int i2 = -1; float b2 = -3.0e38f;
#pragma unroll
    for (int e = 0; e < E_; e++) if (e != i1 && sel[e] > b2) { b2 = sel[e]; i2 = e; }
    float g1 = sv[i1], g2 = sv[i2];
    float denom = g1 + g2;
    float ga, gb;
    if (denom > 1e-9f) { ga = g1 / (denom + 1e-9f); gb = g2 / (denom + 1e-9f); }
    else               { ga = 0.5f; gb = 0.5f; }
    idx[n * 2] = i1; idx[n * 2 + 1] = i2;
    gates[n * 2] = ga; gates[n * 2 + 1] = gb;
}

// Deterministic per-expert token list build (block-scan, no atomics)
__global__ __launch_bounds__(256)
void build_lists(const int* __restrict__ idx, int* __restrict__ list,
                 int* __restrict__ posOf, int* __restrict__ counts) {
    __shared__ int sdata[256];
    __shared__ int base;
    int e = blockIdx.x;
    if (threadIdx.x == 0) base = 0;
    __syncthreads();
    for (int c0 = 0; c0 < NTOK; c0 += 256) {
        int n = c0 + threadIdx.x;
        int k = -1;
        if (idx[n * 2] == e) k = 0;
        else if (idx[n * 2 + 1] == e) k = 1;
        int f = (k >= 0) ? 1 : 0;
        sdata[threadIdx.x] = f;
        __syncthreads();
        for (int o = 1; o < 256; o <<= 1) {
            int v = sdata[threadIdx.x];
            int u = (threadIdx.x >= (unsigned)o) ? sdata[threadIdx.x - o] : 0;
            __syncthreads();
            sdata[threadIdx.x] = v + u;
            __syncthreads();
        }
        int excl = sdata[threadIdx.x] - f;
        if (k >= 0) {
            int pos = base + excl;
            list[e * NTOK + pos] = n;
            posOf[n * 2 + k] = pos;
        }
        __syncthreads();
        if (threadIdx.x == 0) base += sdata[255];
        __syncthreads();
    }
    if (threadIdx.x == 0) counts[e] = base;
}

__global__ void offsets_kernel(const int* __restrict__ counts, int* __restrict__ offs) {
    if (threadIdx.x == 0 && blockIdx.x == 0) {
        int a = 0;
        for (int e = 0; e < E_; e++) { offs[e] = a; a += counts[e]; }
    }
}

// out[n,c] += g0*ybuf[slot0,c] + g1*ybuf[slot1,c]
__global__ __launch_bounds__(256)
void moe_combine(float* __restrict__ out, const float* __restrict__ ybuf,
                 const int* __restrict__ idx, const float* __restrict__ gates,
                 const int* __restrict__ pos, const int* __restrict__ offs) {
    int i = blockIdx.x * 256 + threadIdx.x;   // NTOK*C
    int n = i >> 10;
    int c = i & 1023;
    int e0 = idx[n * 2], e1 = idx[n * 2 + 1];
    int s0 = offs[e0] + pos[n * 2];
    int s1 = offs[e1] + pos[n * 2 + 1];
    out[i] += gates[n * 2]     * ybuf[(size_t)s0 * C_ + c]
            + gates[n * 2 + 1] * ybuf[(size_t)s1 * C_ + c];
}

// =====================================================================
// Host orchestration
// =====================================================================
extern "C" void kernel_launch(void* const* d_in, const int* in_sizes, int n_in,
                              void* d_out, int out_size, void* d_ws, size_t ws_size,
                              hipStream_t stream) {
    const float* x       = (const float*)d_in[0];
    const float* t_emb   = (const float*)d_in[1];
    const float* ada1_w  = (const float*)d_in[2];
    const float* ada1_b  = (const float*)d_in[3];
    const float* ada2_w  = (const float*)d_in[4];
    const float* ada2_b  = (const float*)d_in[5];
    const float* wq      = (const float*)d_in[6];
    const float* wk      = (const float*)d_in[7];
    const float* wv      = (const float*)d_in[8];
    const float* wo      = (const float*)d_in[9];
    const float* sw1     = (const float*)d_in[10];
    const float* sw3     = (const float*)d_in[11];
    const float* sw2     = (const float*)d_in[12];
    const float* re_w1   = (const float*)d_in[13];
    const float* re_w2   = (const float*)d_in[14];
    const float* rout_w  = (const float*)d_in[15];
    const float* rout_b  = (const float*)d_in[16];
    float* out = (float*)d_out;

    // ---- workspace bump allocator (~261 MiB total) ----
    char* ws = (char*)d_ws;
    size_t o = 0;
    auto take = [&](size_t bytes) -> char* {
        char* p = ws + o;
        o += (bytes + 255) & ~(size_t)255;
        return p;
    };
    unsigned short* wqT  = (unsigned short*)take((size_t)1024 * 1024 * 2);
    unsigned short* wkT  = (unsigned short*)take((size_t)1024 * 256 * 2);
    unsigned short* wvT  = (unsigned short*)take((size_t)1024 * 256 * 2);
    unsigned short* woT  = (unsigned short*)take((size_t)1024 * 1024 * 2);
    unsigned short* sw1T = (unsigned short*)take((size_t)2048 * 1024 * 2);
    unsigned short* sw3T = (unsigned short*)take((size_t)2048 * 1024 * 2);
    unsigned short* sw2T = (unsigned short*)take((size_t)1024 * 2048 * 2);
    unsigned short* rw1T = (unsigned short*)take((size_t)E_ * 1024 * 1024 * 2);
    unsigned short* rw2T = (unsigned short*)take((size_t)E_ * 1024 * 1024 * 2);
    float* s1    = (float*)take((size_t)B_ * 2048 * 4);
    float* s2    = (float*)take((size_t)B_ * 2048 * 4);
    float* hln   = (float*)take((size_t)NTOK * C_ * 4);        // f32 LN out (router)
    unsigned short* hbf = (unsigned short*)take((size_t)NTOK * C_ * 2); // bf16 LN out
    float* xqb   = (float*)take((size_t)NTOK * 1024 * 4);
    float* xkb   = (float*)take((size_t)NTOK * 256 * 4);
    float* xvb   = (float*)take((size_t)NTOK * 256 * 4);
    unsigned short* qbf = (unsigned short*)take((size_t)B_ * H_ * T_ * HD_ * 2);
    unsigned short* kbf = (unsigned short*)take((size_t)B_ * KVH_ * T_ * HD_ * 2);
    unsigned short* vtb = (unsigned short*)take((size_t)B_ * KVH_ * HD_ * T_ * 2);
    unsigned short* ybfA = (unsigned short*)take((size_t)NTOK * 1024 * 2); // attn out bf16
    float* x1    = (float*)take((size_t)NTOK * C_ * 4);
    float* abuf  = (float*)take((size_t)NTOK * SH_H_ * 4);
    float* bbuf  = (float*)take((size_t)NTOK * SH_H_ * 4);
    unsigned short* midbf = (unsigned short*)take((size_t)NTOK * SH_H_ * 2);
    float* sbuf  = (float*)take((size_t)NTOK * E_ * 4);
    int*   idxb  = (int*)take((size_t)NTOK * 2 * 4);
    float* gatb  = (float*)take((size_t)NTOK * 2 * 4);
    int*   posb  = (int*)take((size_t)NTOK * 2 * 4);
    int*   listb = (int*)take((size_t)E_ * NTOK * 4);
    int*   cntb  = (int*)take(64);
    int*   offb  = (int*)take(64);
    unsigned short* hbuf = (unsigned short*)take((size_t)NTOK * 2 * RE_H_ * 2); // bf16
    float* ybuf  = (float*)take((size_t)NTOK * 2 * C_ * 4);
    (void)ws_size; (void)in_sizes; (void)n_in; (void)out_size;

    dim3 tb(32, 8);
    // ---- weight convert + transpose to bf16 [N,K] ----
    w_to_bf16t<<<dim3(32, 32, 1), tb, 0, stream>>>(wq, wqT, 1024, 1024);
    w_to_bf16t<<<dim3(8,  32, 1), tb, 0, stream>>>(wk, wkT, 1024, 256);
    w_to_bf16t<<<dim3(8,  32, 1), tb, 0, stream>>>(wv, wvT, 1024, 256);
    w_to_bf16t<<<dim3(32, 32, 1), tb, 0, stream>>>(wo, woT, 1024, 1024);
    w_to_bf16t<<<dim3(64, 32, 1), tb, 0, stream>>>(sw1, sw1T, 1024, 2048);
    w_to_bf16t<<<dim3(64, 32, 1), tb, 0, stream>>>(sw3, sw3T, 1024, 2048);
    w_to_bf16t<<<dim3(32, 64, 1), tb, 0, stream>>>(sw2, sw2T, 2048, 1024);
    w_to_bf16t<<<dim3(32, 32, E_), tb, 0, stream>>>(re_w1, rw1T, 1024, 1024);
    w_to_bf16t<<<dim3(32, 32, E_), tb, 0, stream>>>(re_w2, rw2T, 1024, 1024);

    // ---- AdaLN modulation + h1 ----
    timemod_kernel<<<16, 256, 0, stream>>>(t_emb, ada1_w, ada1_b, s1);
    timemod_kernel<<<16, 256, 0, stream>>>(t_emb, ada2_w, ada2_b, s2);
    adaln_kernel<<<NTOK, 256, 0, stream>>>(x, s1, hln, hbf);

    // ---- QKV projections (WMMA; 256-row blocks -> grid.y = 16) ----
    wmma_gemm_t<0, false, false, false, false><<<dim3(16, 16, 1), 256, 0, stream>>>(
        hbf, wqT, xqb, nullptr, nullptr, nullptr, nullptr, NTOK, 1024, 1024, 0);
    wmma_gemm_t<0, false, false, false, false><<<dim3(4, 16, 1), 256, 0, stream>>>(
        hbf, wkT, xkb, nullptr, nullptr, nullptr, nullptr, NTOK, 256, 1024, 0);
    wmma_gemm_t<0, false, false, false, false><<<dim3(4, 16, 1), 256, 0, stream>>>(
        hbf, wvT, xvb, nullptr, nullptr, nullptr, nullptr, NTOK, 256, 1024, 0);

    // ---- RoPE + bf16 pack ----
    rope_pack_q<<<(B_ * T_ * H_ * HD_) / 256, 256, 0, stream>>>(xqb, qbf);
    rope_pack_k<<<(B_ * T_ * KVH_ * HD_) / 256, 256, 0, stream>>>(xkb, kbf);
    v_pack_t   <<<(B_ * T_ * KVH_ * HD_) / 256, 256, 0, stream>>>(xvb, vtb);

    // ---- flash attention (WMMA), bf16 output ----
    flash_attn<<<dim3(T_ / 16, H_, B_), 32, 0, stream>>>(qbf, kbf, vtb, ybfA);

    // ---- out-proj + residual: x1 = x + y@wo ----
    wmma_gemm_t<0, true, false, false, false><<<dim3(16, 16, 1), 256, 0, stream>>>(
        ybfA, woT, x1, x, nullptr, nullptr, nullptr, NTOK, 1024, 1024, 0);

    // ---- second AdaLN -> h2 (reuse hln/hbf) ----
    adaln_kernel<<<NTOK, 256, 0, stream>>>(x1, s2, hln, hbf);

    // ---- shared SwiGLU FFN ----
    wmma_gemm_t<0, false, false, false, false><<<dim3(32, 16, 1), 256, 0, stream>>>(
        hbf, sw1T, abuf, nullptr, nullptr, nullptr, nullptr, NTOK, SH_H_, 1024, 0);
    wmma_gemm_t<0, false, false, false, false><<<dim3(32, 16, 1), 256, 0, stream>>>(
        hbf, sw3T, bbuf, nullptr, nullptr, nullptr, nullptr, NTOK, SH_H_, 1024, 0);
    silu_mul<<<(NTOK * SH_H_) / 256, 256, 0, stream>>>(abuf, bbuf, midbf);
    // out = x1 + mid@sw2 (routed experts added later)
    wmma_gemm_t<0, true, false, false, false><<<dim3(16, 16, 1), 256, 0, stream>>>(
        midbf, sw2T, out, x1, nullptr, nullptr, nullptr, NTOK, 1024, SH_H_, 0);

    // ---- MoE routing (deterministic, no atomics) ----
    router_scores<<<(NTOK * E_) / 256, 256, 0, stream>>>(hln, rout_w, sbuf);
    topk_kernel<<<NTOK / 256, 256, 0, stream>>>(sbuf, rout_b, idxb, gatb);
    build_lists<<<E_, 256, 0, stream>>>(idxb, listb, posb, cntb);
    offsets_kernel<<<1, 1, 0, stream>>>(cntb, offb);

    // ---- routed experts: gathered GEMM1 (GELU -> bf16) -> compact GEMM2 ----
    wmma_gemm_t<1, false, true, true, true><<<dim3(RE_H_ / 64, NTOK / 256, E_), 256, 0, stream>>>(
        hbf, rw1T, hbuf, nullptr, listb, cntb, offb, NTOK, RE_H_, 1024,
        (long long)1024 * 1024);
    wmma_gemm_t<0, false, true, false, false><<<dim3(C_ / 64, NTOK / 256, E_), 256, 0, stream>>>(
        hbuf, rw2T, ybuf, nullptr, nullptr, cntb, offb, NTOK, C_, RE_H_,
        (long long)1024 * 1024);

    // ---- gate-weighted combine into out ----
    moe_combine<<<(NTOK * C_) / 256, 256, 0, stream>>>(out, ybuf, idxb, gatb, posb, offb);
}
